// StackedEarlyExit_721554505902
// MI455X (gfx1250) — matrix operand order repaired and checked by reading, not compiled, not run on hardware
//
#include <hip/hip_runtime.h>
#include <hip/hip_bf16.h>

typedef __attribute__((ext_vector_type(16))) _Float16 v16h;
typedef __attribute__((ext_vector_type(8)))  float    v8f;

#define NUM_CLASSES 30
#define NUM_EXITS   3
#define HIDDEN      256
#define COMBOS      27000      // 30^3
#define TILES       1688       // ceil(27000/16)
#define WAVES_PER_BLOCK 8
#define LUT_BLOCKS  211        // 1688 / 8 exactly
#define ROWS_PER_BLOCK 256

__device__ __forceinline__ void load8(const float* __restrict__ p, float* d) {
  float4 u = *(const float4*)p;
  float4 v = *(const float4*)(p + 4);
  d[0] = u.x; d[1] = u.y; d[2] = u.z; d[3] = u.w;
  d[4] = v.x; d[5] = v.y; d[6] = v.z; d[7] = v.w;
}

// ---------------------------------------------------------------------------
// Kernel 1: build LUT[27000][30] = relu(W1[i0]+W1[30+i1]+W1[60+i2]+b1) @ W2 + b2
// One wave handles 16 combos (M=16); N = 30 padded into two 16-wide WMMA tiles;
// K = 256 consumed 32 at a time with v_wmma_f32_16x16x32_f16.
// ---------------------------------------------------------------------------
__global__ void __launch_bounds__(256) build_lut_kernel(
    const float* __restrict__ W1, const float* __restrict__ b1,
    const float* __restrict__ W2, const float* __restrict__ b2,
    float* __restrict__ lut) {
  // B fragments in native WMMA B layout: [kb:8][lane:32][tile:2][f:16] halfs
  __shared__ __align__(32) _Float16 Bfrag[8 * 32 * 2 * 16];  // 16 KB
  __shared__ float b1s[HIDDEN];                              // 1 KB

  const int tid = threadIdx.x;
  if (tid < HIDDEN) b1s[tid] = b1[tid];
  // Stage W2 (f32 -> f16) directly in B-fragment order:
  // lane holds column n = tile*16 + (lane&15); K {0..7,16..23} (lanes 0-15) / {8..15,24..31} (lanes 16-31).
  for (int i = tid; i < 8192; i += 256) {
    const int f    = i & 15;
    const int tile = (i >> 4) & 1;
    const int ln   = (i >> 5) & 31;
    const int kb   = i >> 10;
    const int k = kb * 32 + ((ln & 16) ? 8 : 0) + (f & 7) + ((f & 8) ? 16 : 0);
    const int n = tile * 16 + (ln & 15);
    const float v = (n < NUM_CLASSES) ? W2[k * NUM_CLASSES + n] : 0.0f;
    Bfrag[i] = (_Float16)v;
  }
  __syncthreads();

  const int lane  = tid & 31;
  const int wid   = tid >> 5;
  const int tile  = blockIdx.x * WAVES_PER_BLOCK + wid;   // < 1688
  const int cbase = tile * 16;

  // Combo whose hidden-row this lane supplies for the A matrix (M = lane&15).
  int c = cbase + (lane & 15);
  if (c > COMBOS - 1) c = COMBOS - 1;                      // tail tile: clamp
  const int i0  = c / 900;
  const int rem = c - i0 * 900;
  const int i1  = rem / 30;
  const int i2  = rem - i1 * 30;
  const float* __restrict__ wa = W1 + (size_t)i0 * HIDDEN;
  const float* __restrict__ wb = W1 + (size_t)(30 + i1) * HIDDEN;
  const float* __restrict__ wc = W1 + (size_t)(60 + i2) * HIDDEN;
  // 16-bit A layout: lanes 0-15 cover K {0..7,16..23}+32*kb, lanes 16-31 {8..15,24..31}
  const int hiBase = (lane & 16) ? 8 : 0;

  v8f acc0 = {};
  v8f acc1 = {};
  #pragma unroll
  for (int kb = 0; kb < HIDDEN / 32; ++kb) {
    const int k0 = kb * 32 + hiBase;        // halfs 0..7
    const int k1 = kb * 32 + 16 + hiBase;   // halfs 8..15
    float s[8], t[8];
    v16h a;
    // halfs 0..7: hidden = relu(wa + wb + wc + b1)
    load8(wa + k0, s); load8(wb + k0, t);
    #pragma unroll
    for (int j = 0; j < 8; ++j) s[j] += t[j];
    load8(wc + k0, t);
    #pragma unroll
    for (int j = 0; j < 8; ++j) s[j] += t[j];
    load8(b1s + k0, t);
    #pragma unroll
    for (int j = 0; j < 8; ++j) a[j] = (_Float16)fmaxf(s[j] + t[j], 0.0f);
    // halfs 8..15
    load8(wa + k1, s); load8(wb + k1, t);
    #pragma unroll
    for (int j = 0; j < 8; ++j) s[j] += t[j];
    load8(wc + k1, t);
    #pragma unroll
    for (int j = 0; j < 8; ++j) s[j] += t[j];
    load8(b1s + k1, t);
    #pragma unroll
    for (int j = 0; j < 8; ++j) a[8 + j] = (_Float16)fmaxf(s[j] + t[j], 0.0f);

    const v16h b0  = *(const v16h*)&Bfrag[kb * 1024 + lane * 32];
    const v16h b1v = *(const v16h*)&Bfrag[kb * 1024 + lane * 32 + 16];
    acc0 = __builtin_amdgcn_wmma_f32_16x16x32_f16(false, a, false, b0,
                                                  (short)0, acc0, false, false);
    acc1 = __builtin_amdgcn_wmma_f32_16x16x32_f16(false, a, false, b1v,
                                                  (short)0, acc1, false, false);
  }

  // D layout: VGPR v -> M = v + 8*(lane/16), N = lane&15 (+16 for tile 1).
  const int nA = lane & 15;
  const int nB = 16 + nA;
  const int mBase = cbase + ((lane & 16) ? 8 : 0);
  const float b2a = b2[nA];
  const float b2b = (nB < NUM_CLASSES) ? b2[nB] : 0.0f;
  #pragma unroll
  for (int v = 0; v < 8; ++v) {
    const int cc = mBase + v;
    if (cc < COMBOS) {
      lut[(size_t)cc * NUM_CLASSES + nA] = acc0[v] + b2a;
      if (nB < NUM_CLASSES) lut[(size_t)cc * NUM_CLASSES + nB] = acc1[v] + b2b;
    }
  }
}

// ---------------------------------------------------------------------------
// Kernel 2: streaming pass. Coalesced b128 stage of 256 rows -> LDS,
// per-row argmax x3 (first-occurrence semantics), LUT gather, coalesced store.
// ---------------------------------------------------------------------------
__global__ void __launch_bounds__(256) apply_kernel(
    const float* __restrict__ preds, const float* __restrict__ lut,
    float* __restrict__ out, int nRows) {
  __shared__ __align__(16) float lds[ROWS_PER_BLOCK * 90];  // 92,160 B

  const int tid = threadIdx.x;
  const long long R = (long long)blockIdx.x * ROWS_PER_BLOCK;
  if (R >= nRows) return;

  // Stage 256 rows (fully coalesced 16B loads; base = blockIdx*92160 B, 16B aligned)
  const float4* __restrict__ src4 = (const float4*)(preds + R * 90);
  float4* dst4 = (float4*)lds;
  #pragma unroll 4
  for (int i = tid; i < ROWS_PER_BLOCK * 90 / 4; i += 256) dst4[i] = src4[i];
  __syncthreads();

  // Per-row argmax over each exit head (strict > => first occurrence, as jnp.argmax)
  const float* p = lds + tid * 90;   // stride 90 words mod 64 banks: conflict-free
  int combo = 0;
  #pragma unroll
  for (int e = 0; e < NUM_EXITS; ++e) {
    const float2* pe = (const float2*)(p + e * NUM_CLASSES);
    float best = -3.4028235e38f;
    int bi = 0;
    #pragma unroll
    for (int j = 0; j < NUM_CLASSES / 2; ++j) {
      const float2 v = pe[j];
      if (v.x > best) { best = v.x; bi = 2 * j; }
      if (v.y > best) { best = v.y; bi = 2 * j + 1; }
    }
    combo = combo * NUM_CLASSES + bi;
  }
  __syncthreads();   // everyone done reading preds before LDS reuse

  // Gather LUT row (L2-resident, 3.24 MB) into LDS for coalesced output
  const float2* __restrict__ lrow = (const float2*)(lut + (size_t)combo * NUM_CLASSES);
  float2* orow = (float2*)(lds + tid * NUM_CLASSES);
  #pragma unroll
  for (int j = 0; j < NUM_CLASSES / 2; ++j) orow[j] = lrow[j];
  __syncthreads();

  // Coalesced 16B stores (base = blockIdx*30720 B, 16B aligned)
  float4* __restrict__ gout = (float4*)(out + R * NUM_CLASSES);
  const float4* lds4 = (const float4*)lds;
  #pragma unroll 4
  for (int i = tid; i < ROWS_PER_BLOCK * NUM_CLASSES / 4; i += 256) gout[i] = lds4[i];
}

// ---------------------------------------------------------------------------
extern "C" void kernel_launch(void* const* d_in, const int* in_sizes, int n_in,
                              void* d_out, int out_size, void* d_ws, size_t ws_size,
                              hipStream_t stream) {
  const float* preds = (const float*)d_in[0];  // [B, 3, 30]
  const float* W1    = (const float*)d_in[1];  // [90, 256]
  const float* b1    = (const float*)d_in[2];  // [256]
  const float* W2    = (const float*)d_in[3];  // [256, 30]
  const float* b2    = (const float*)d_in[4];  // [30]
  float* out = (float*)d_out;
  float* lut = (float*)d_ws;                   // needs 27000*30*4 = 3.24 MB scratch

  const int nRows = in_sizes[0] / (NUM_EXITS * NUM_CLASSES);  // 1,048,576

  // 1) Build the 27,000-entry output LUT with WMMA (negligible: ~0.44 GFLOP).
  build_lut_kernel<<<LUT_BLOCKS, 256, 0, stream>>>(W1, b1, W2, b2, lut);

  // 2) Memory-bound streaming pass: ~480 MB total -> ~21 us at 23.3 TB/s.
  const int blocks = (nRows + ROWS_PER_BLOCK - 1) / ROWS_PER_BLOCK;  // 4096
  apply_kernel<<<blocks, 256, 0, stream>>>(preds, lut, out, nRows);
}
